// ClipLoss_72765335929211
// MI455X (gfx1250) — matrix operand order repaired and checked by reading, not compiled, not run on hardware
//
#include <hip/hip_runtime.h>
#include <math.h>

#define B   512
#define C   3
#define NP  49
#define NT  76
#define D   768
#define EPS 0.1f
#define THRESH 0.01f
#define MAX_ITER 100

typedef __attribute__((ext_vector_type(16))) _Float16 v16h;
typedef __attribute__((ext_vector_type(8)))  _Float16 v8h;
typedef __attribute__((ext_vector_type(8)))  float    v8f;

#define ASTR 80  // LDS row stride in halfs (160B, multiple of 16B for v8h loads)

#if __has_builtin(__builtin_amdgcn_sched_barrier)
#define SCHED_FENCE() __builtin_amdgcn_sched_barrier(0)
#else
#define SCHED_FENCE()
#endif

// ---------------------------------------------------------------------------
// Kernel A: per-row inverse L2 norms of local image/text features.
// One wave (32 lanes) per row of length D=768.
// ---------------------------------------------------------------------------
__global__ __launch_bounds__(256) void norms_kernel(
    const float* __restrict__ li, const float* __restrict__ lt,
    float* __restrict__ inv_img, float* __restrict__ inv_txt) {
  const int NIMG = B * C * NP;
  const int NTOT = NIMG + B * C * NT;
  int wave = blockIdx.x * (blockDim.x >> 5) + (threadIdx.x >> 5);
  int lane = threadIdx.x & 31;
  if (wave >= NTOT) return;
  const float* src = (wave < NIMG) ? (li + (size_t)wave * D)
                                   : (lt + (size_t)(wave - NIMG) * D);
  float ss = 0.f;
  #pragma unroll
  for (int k = 0; k < D / 32; ++k) {
    float x = src[k * 32 + lane];
    ss += x * x;
  }
  #pragma unroll
  for (int off = 16; off; off >>= 1) ss += __shfl_xor(ss, off, 32);
  if (lane == 0) {
    float inv = 1.0f / fmaxf(sqrtf(ss), 1e-12f);
    if (wave < NIMG) inv_img[wave] = inv;
    else             inv_txt[wave - NIMG] = inv;
  }
}

// ---------------------------------------------------------------------------
// Kernel B: sim[b,c] = norm(li[b,c]) @ norm(lt[b,c])^T  via WMMA f16.
// One block (4 waves) per (b,c). 49x76 output padded to 64x80 tiles.
// A sched_barrier pins all ds_loads of a K-step before the WMMA burst so
// DScnt drains once per K-step instead of once per WMMA.
// ---------------------------------------------------------------------------
__global__ __launch_bounds__(128) void sim_kernel(
    const float* __restrict__ li, const float* __restrict__ lt,
    const float* __restrict__ inv_img, const float* __restrict__ inv_txt,
    float* __restrict__ sim) {
  __shared__ __align__(16) _Float16 As[64 * ASTR];
  __shared__ __align__(16) _Float16 Bs[80 * ASTR];
  int bc   = blockIdx.x;
  int tid  = threadIdx.x;
  int wv   = tid >> 5;
  int lane = tid & 31;
  const float* gi = li + (size_t)bc * NP * D;
  const float* gt = lt + (size_t)bc * NT * D;
  const float* ii = inv_img + bc * NP;
  const float* it = inv_txt + bc * NT;

  // zero once (covers pad rows 49..63 / 76..79 for the whole K loop)
  for (int i = tid; i < 64 * ASTR; i += 128) As[i] = (_Float16)0.f;
  for (int i = tid; i < 80 * ASTR; i += 128) Bs[i] = (_Float16)0.f;

  v8f acc[5] = {};
  for (int kc = 0; kc < D; kc += 64) {
    __syncthreads();
    for (int i = tid; i < NP * 64; i += 128) {
      int m = i >> 6, k = i & 63;
      As[m * ASTR + k] = (_Float16)(gi[(size_t)m * D + kc + k] * ii[m]);
      if (kc + 64 < D)  // prefetch next K-chunk (speculative, cacheline-wide)
        __builtin_prefetch(&gi[(size_t)m * D + kc + 64 + k], 0, 0);
    }
    for (int i = tid; i < NT * 64; i += 128) {
      int n = i >> 6, k = i & 63;
      Bs[n * ASTR + k] = (_Float16)(gt[(size_t)n * D + kc + k] * it[n]);
      if (kc + 64 < D)
        __builtin_prefetch(&gt[(size_t)n * D + kc + 64 + k], 0, 0);
    }
    __syncthreads();
    #pragma unroll
    for (int ks = 0; ks < 64; ks += 32) {
      // A fragment: lanes 0-15 -> M=lane, K ks+0..7 & ks+16..23;
      //             lanes 16-31 -> M=lane-16, K ks+8..15 & ks+24..31.
      int am  = 16 * wv + (lane & 15);
      int akb = ks + ((lane >> 4) << 3);
      v8h alo = *(const v8h*)(&As[am * ASTR + akb]);
      v8h ahi = *(const v8h*)(&As[am * ASTR + akb + 16]);
      v16h a;
      #pragma unroll
      for (int q = 0; q < 8; ++q) { a[q] = alo[q]; a[8 + q] = ahi[q]; }
      int bn0 = lane & 15;
      int bkb = ks + ((lane >> 4) << 3);
      v16h bf[5];
      #pragma unroll
      for (int j = 0; j < 5; ++j) {
        int bn = 16 * j + bn0;
        v8h blo = *(const v8h*)(&Bs[bn * ASTR + bkb]);
        v8h bhi = *(const v8h*)(&Bs[bn * ASTR + bkb + 16]);
        #pragma unroll
        for (int q = 0; q < 8; ++q) { bf[j][q] = blo[q]; bf[j][8 + q] = bhi[q]; }
      }
      SCHED_FENCE();  // keep all ds_loads above the WMMA burst
      #pragma unroll
      for (int j = 0; j < 5; ++j)
        acc[j] = __builtin_amdgcn_wmma_f32_16x16x32_f16(
            false, a, false, bf[j], (short)0, acc[j], false, false);
      SCHED_FENCE();
    }
  }
  float* so = sim + (size_t)bc * NP * NT;
  int mbase = 16 * wv + ((lane >> 4) << 3);
  int nn    = lane & 15;
  #pragma unroll
  for (int j = 0; j < 5; ++j) {
    int n = 16 * j + nn;
    if (n < NT) {
      #pragma unroll
      for (int v = 0; v < 8; ++v) {
        int m = mbase + v;
        if (m < NP) so[m * NT + n] = acc[j][v];
      }
    }
  }
}

// ---------------------------------------------------------------------------
// Kernel C: Sinkhorn per (b,ch) fully in LDS; writes partial OT loss.
// ---------------------------------------------------------------------------
__global__ __launch_bounds__(128) void sinkhorn_kernel(
    const float* __restrict__ sim, float* __restrict__ ot_partial) {
  __shared__ float Ksh[NP * NT];
  __shared__ float Ssh[NP * NT];
  __shared__ float rS[NP];
  __shared__ float cS[NT];
  __shared__ float red[128];
  int bc = blockIdx.x;
  int tid = threadIdx.x;
  const float* s = sim + (size_t)bc * NP * NT;
  for (int i = tid; i < NP * NT; i += 128) {
    float v = s[i];
    Ssh[i] = v;
    Ksh[i] = __expf((v - 1.0f) * (1.0f / EPS));
  }
  if (tid < NP) rS[tid] = 1.0f;
  if (tid < NT) cS[tid] = 1.0f;
  __syncthreads();
  const float u = 1.0f / NP, vv = 1.0f / NT;
  for (int it = 0; it < MAX_ITER; ++it) {
    float d = 0.f;
    if (tid < NP) {
      float dot = 0.f;
      for (int n = 0; n < NT; ++n) dot += Ksh[tid * NT + n] * cS[n];
      float rn = u / dot;
      d = fabsf(rn - rS[tid]);
      rS[tid] = rn;
    }
    red[tid] = d;
    __syncthreads();
    #pragma unroll
    for (int st = 64; st; st >>= 1) {
      if (tid < st) red[tid] += red[tid + st];
      __syncthreads();
    }
    float err = red[0] * (1.0f / NP);
    if (tid < NT) {
      float dot = 0.f;
      for (int m = 0; m < NP; ++m) dot += Ksh[m * NT + tid] * rS[m];
      cS[tid] = vv / dot;
    }
    __syncthreads();
    if (err < THRESH) break;  // uniform across block
  }
  float part = 0.f;
  for (int i = tid; i < NP * NT; i += 128)
    part += rS[i / NT] * cS[i % NT] * Ksh[i] * Ssh[i];
  red[tid] = part;
  __syncthreads();
  #pragma unroll
  for (int st = 64; st; st >>= 1) {
    if (tid < st) red[tid] += red[tid + st];
    __syncthreads();
  }
  if (tid == 0) ot_partial[bc] = red[0];
}

// ---------------------------------------------------------------------------
// Kernel D: logits = (scale/C) * image[B,C*D] @ text[B,C*D]^T via WMMA f16.
// 8x8 grid of 64x64 tiles, K = 2304.
// ---------------------------------------------------------------------------
__global__ __launch_bounds__(128) void logits_kernel(
    const float* __restrict__ img, const float* __restrict__ txt,
    const float* __restrict__ scale_p, float* __restrict__ logits) {
  __shared__ __align__(16) _Float16 As[64 * ASTR];
  __shared__ __align__(16) _Float16 Bs[64 * ASTR];
  const int K = C * D;  // 2304
  int bm = blockIdx.x, bn = blockIdx.y;
  int tid = threadIdx.x, wv = tid >> 5, lane = tid & 31;
  const float* ga = img + (size_t)bm * 64 * K;
  const float* gb = txt + (size_t)bn * 64 * K;
  v8f acc[4] = {};
  for (int kc = 0; kc < K; kc += 64) {
    __syncthreads();
    for (int i = tid; i < 64 * 64; i += 128) {
      int r = i >> 6, k = i & 63;
      As[r * ASTR + k] = (_Float16)ga[(size_t)r * K + kc + k];
      Bs[r * ASTR + k] = (_Float16)gb[(size_t)r * K + kc + k];
      if (kc + 64 < K) {
        __builtin_prefetch(&ga[(size_t)r * K + kc + 64 + k], 0, 0);
        __builtin_prefetch(&gb[(size_t)r * K + kc + 64 + k], 0, 0);
      }
    }
    __syncthreads();
    #pragma unroll
    for (int ks = 0; ks < 64; ks += 32) {
      int am  = 16 * wv + (lane & 15);
      int akb = ks + ((lane >> 4) << 3);
      v8h alo = *(const v8h*)(&As[am * ASTR + akb]);
      v8h ahi = *(const v8h*)(&As[am * ASTR + akb + 16]);
      v16h a;
      #pragma unroll
      for (int q = 0; q < 8; ++q) { a[q] = alo[q]; a[8 + q] = ahi[q]; }
      int bn0 = lane & 15;
      int bkb = ks + ((lane >> 4) << 3);
      v16h bf[4];
      #pragma unroll
      for (int j = 0; j < 4; ++j) {
        int bcol = 16 * j + bn0;
        v8h blo = *(const v8h*)(&Bs[bcol * ASTR + bkb]);
        v8h bhi = *(const v8h*)(&Bs[bcol * ASTR + bkb + 16]);
        #pragma unroll
        for (int q = 0; q < 8; ++q) { bf[j][q] = blo[q]; bf[j][8 + q] = bhi[q]; }
      }
      SCHED_FENCE();  // keep all ds_loads above the WMMA burst
      #pragma unroll
      for (int j = 0; j < 4; ++j)
        acc[j] = __builtin_amdgcn_wmma_f32_16x16x32_f16(
            false, a, false, bf[j], (short)0, acc[j], false, false);
      SCHED_FENCE();
    }
  }
  float sc = scale_p[0] / (float)C;
  int mg0 = bm * 64 + 16 * wv + ((lane >> 4) << 3);
  int ng0 = bn * 64 + (lane & 15);
  #pragma unroll
  for (int j = 0; j < 4; ++j)
    #pragma unroll
    for (int v = 0; v < 8; ++v)
      logits[(size_t)(mg0 + v) * B + ng0 + 16 * j] = acc[j][v] * sc;
}

// ---------------------------------------------------------------------------
// Kernel E: per-row and per-column (logsumexp - diag) cross-entropy terms.
// ---------------------------------------------------------------------------
__global__ __launch_bounds__(128) void ce_kernel(
    const float* __restrict__ logits,
    float* __restrict__ ce_rows, float* __restrict__ ce_cols) {
  __shared__ float red[128];
  int b = blockIdx.x, tid = threadIdx.x;
  // ---- row b ----
  float mx = -INFINITY;
  for (int j = tid; j < B; j += 128) mx = fmaxf(mx, logits[(size_t)b * B + j]);
  red[tid] = mx; __syncthreads();
  #pragma unroll
  for (int st = 64; st; st >>= 1) { if (tid < st) red[tid] = fmaxf(red[tid], red[tid + st]); __syncthreads(); }
  mx = red[0]; __syncthreads();
  float se = 0.f;
  for (int j = tid; j < B; j += 128) se += __expf(logits[(size_t)b * B + j] - mx);
  red[tid] = se; __syncthreads();
  #pragma unroll
  for (int st = 64; st; st >>= 1) { if (tid < st) red[tid] += red[tid + st]; __syncthreads(); }
  if (tid == 0) ce_rows[b] = mx + __logf(red[0]) - logits[(size_t)b * B + b];
  __syncthreads();
  // ---- column b ----
  mx = -INFINITY;
  for (int j = tid; j < B; j += 128) mx = fmaxf(mx, logits[(size_t)j * B + b]);
  red[tid] = mx; __syncthreads();
  #pragma unroll
  for (int st = 64; st; st >>= 1) { if (tid < st) red[tid] = fmaxf(red[tid], red[tid + st]); __syncthreads(); }
  mx = red[0]; __syncthreads();
  se = 0.f;
  for (int j = tid; j < B; j += 128) se += __expf(logits[(size_t)j * B + b] - mx);
  red[tid] = se; __syncthreads();
  #pragma unroll
  for (int st = 64; st; st >>= 1) { if (tid < st) red[tid] += red[tid + st]; __syncthreads(); }
  if (tid == 0) ce_cols[b] = mx + __logf(red[0]) - logits[(size_t)b * B + b];
}

// ---------------------------------------------------------------------------
// Kernel F: deterministic final reduction -> scalar loss.
// ---------------------------------------------------------------------------
__global__ __launch_bounds__(256) void final_kernel(
    const float* __restrict__ ot_partial, const float* __restrict__ ce_rows,
    const float* __restrict__ ce_cols, float* __restrict__ out) {
  __shared__ float red[256];
  int tid = threadIdx.x;
  float ot = 0.f;
  for (int i = tid; i < B * C; i += 256) ot += ot_partial[i];
  float cr = 0.f, cc = 0.f;
  for (int i = tid; i < B; i += 256) { cr += ce_rows[i]; cc += ce_cols[i]; }
  red[tid] = ot + (cr + cc) * (0.5f / (float)B);
  __syncthreads();
  #pragma unroll
  for (int st = 128; st; st >>= 1) {
    if (tid < st) red[tid] += red[tid + st];
    __syncthreads();
  }
  if (tid == 0) out[0] = red[0];
}

// ---------------------------------------------------------------------------
extern "C" void kernel_launch(void* const* d_in, const int* in_sizes, int n_in,
                              void* d_out, int out_size, void* d_ws, size_t ws_size,
                              hipStream_t stream) {
  const float* image = (const float*)d_in[0];  // [B,C,D]
  const float* text  = (const float*)d_in[1];  // [B,C,D]
  const float* scale = (const float*)d_in[2];  // [1]
  const float* li    = (const float*)d_in[3];  // [B,C,NP,D]
  const float* lt    = (const float*)d_in[4];  // [B,C,NT,D]
  float* out = (float*)d_out;

  // workspace layout (floats)
  float* ws = (float*)d_ws;
  const size_t N_INV_I = (size_t)B * C * NP;          // 75264
  const size_t N_INV_T = (size_t)B * C * NT;          // 116736
  const size_t N_SIM   = (size_t)B * C * NP * NT;     // 5720064
  float* inv_img    = ws;
  float* inv_txt    = inv_img + N_INV_I;
  float* sim        = inv_txt + N_INV_T;
  float* logits     = sim + N_SIM;
  float* ot_partial = logits + (size_t)B * B;
  float* ce_rows    = ot_partial + (size_t)B * C;
  float* ce_cols    = ce_rows + B;

  // A: inverse norms (192000 rows, 8 waves/block)
  norms_kernel<<<24000, 256, 0, stream>>>(li, lt, inv_img, inv_txt);
  // B: per-(b,c) 49x76x768 WMMA GEMM
  sim_kernel<<<B * C, 128, 0, stream>>>(li, lt, inv_img, inv_txt, sim);
  // C: Sinkhorn + partial OT loss per (b,c)
  sinkhorn_kernel<<<B * C, 128, 0, stream>>>(sim, ot_partial);
  // D: 512x512x2304 WMMA GEMM for logits
  logits_kernel<<<dim3(8, 8), 128, 0, stream>>>(image, text, scale, logits);
  // E: row/col cross-entropy terms
  ce_kernel<<<B, 128, 0, stream>>>(logits, ce_rows, ce_cols);
  // F: deterministic scalar reduction
  final_kernel<<<1, 256, 0, stream>>>(ot_partial, ce_rows, ce_cols, out);
}
